// EEG_GNN_47502338294010
// MI455X (gfx1250) — matrix operand order repaired and checked by reading, not compiled, not run on hardware
//
#include <hip/hip_runtime.h>

typedef __attribute__((ext_vector_type(16))) _Float16 v16h;
typedef __attribute__((ext_vector_type(8)))  float    v8f;
typedef int vint4 __attribute__((vector_size(16)));   // matches builtin's V4i

#define EPSV 1e-5f
#define HIDN 128
#define TLEN 500

// ---------------- CDNA5 async global->LDS copy (ASYNCcnt-tracked) ---------
#if defined(__has_builtin)
#if __has_builtin(__builtin_amdgcn_global_load_async_to_lds_b128)
#define HAS_ASYNC_G2L 1
#endif
#endif

__device__ __forceinline__ void async_copy16(void* l, const void* gp) {
#ifdef HAS_ASYNC_G2L
  __builtin_amdgcn_global_load_async_to_lds_b128(
      (__attribute__((address_space(1))) vint4*)(void*)gp,
      (__attribute__((address_space(3))) vint4*)l, 0, 0);
#else
  *(float4*)l = *(const float4*)gp;
#endif
}
__device__ __forceinline__ void wait_async0() {
#if defined(__has_builtin) && __has_builtin(__builtin_amdgcn_s_wait_asynccnt)
  __builtin_amdgcn_s_wait_asynccnt(0);
#else
  asm volatile("s_wait_asynccnt 0" ::: "memory");
#endif
}

// ---------------- WMMA fragment helpers (wave32, per CDNA5 ISA 7.12.2) ----
__device__ __forceinline__ v8f wmma16x16x32(v16h a, v16h b, v8f c) {
  // (neg_a, A, neg_b, B, c_mod, C, reuse_a, reuse_b)
  return __builtin_amdgcn_wmma_f32_16x16x32_f16(false, a, false, b,
                                                (short)0, c, false, false);
}

// A tile 16(M) x 32(K), row-major source with leading dim ld.
// lane 0-15: row M=lane, holds K 0..15 (e=0..15)
// lane 16-31: row M=lane-16, holds K 16..31
__device__ __forceinline__ v16h frag_a_f16(const _Float16* t, int ld) {
  int lane = threadIdx.x & 31, hh = lane >> 4, r = lane & 15;
  v16h o;
#pragma unroll
  for (int e = 0; e < 16; ++e) {
    int k = hh * 16 + e;
    o[e] = t[r * ld + k];
  }
  return o;
}
__device__ __forceinline__ v16h frag_a_f32(const float* t, int ld) {
  int lane = threadIdx.x & 31, hh = lane >> 4, r = lane & 15;
  v16h o;
#pragma unroll
  for (int e = 0; e < 16; ++e) {
    int k = hh * 16 + e;
    o[e] = (_Float16)t[r * ld + k];
  }
  return o;
}
// B tile 32(K) x 16(N), row-major source with leading dim ld.
// lane 0-15: col N=lane, K=0..15 ; lane 16-31: col N=lane-16, K=16..31
__device__ __forceinline__ v16h frag_b_f16(const _Float16* t, int ld) {
  int lane = threadIdx.x & 31, n = lane & 15, kb = (lane >> 4) * 16;
  v16h o;
#pragma unroll
  for (int e = 0; e < 16; ++e) o[e] = t[(kb + e) * ld + n];
  return o;
}
__device__ __forceinline__ v16h frag_b_f32(const float* t, int ld) {
  int lane = threadIdx.x & 31, n = lane & 15, kb = (lane >> 4) * 16;
  v16h o;
#pragma unroll
  for (int e = 0; e < 16; ++e) o[e] = (_Float16)t[(kb + e) * ld + n];
  return o;
}

// ---------------- Kernel 1: fused temporal conv + BN + ReLU + time mean ---
// One block per node (4096 blocks, 8 waves). Conv as GEMM:
// O[128ch, 512t] = W[128,32] @ Xwin[32,512]; epilogue fuses BN/ReLU/mean.
__global__ __launch_bounds__(256) void k_conv(
    const float* __restrict__ x, const float* __restrict__ cw,
    const float* __restrict__ cb, const float* __restrict__ bg,
    const float* __restrict__ bb, const float* __restrict__ brm,
    const float* __restrict__ brv, float* __restrict__ feat) {
  __shared__ __align__(16) float xraw[500];  // raw f32 signal (async-staged)
  __shared__ _Float16 xpad[544];             // xpad[i] = x[i-12], zero padded
  __shared__ _Float16 wk[HIDN * 32];         // weights, K padded 25 -> 32
  __shared__ float sc[HIDN], sh[HIDN];
  int tid = threadIdx.x, node = blockIdx.x;
  // async DMA the node's 2000B time series straight into LDS (125 x 16B)
  if (tid < 125) async_copy16(xraw + tid * 4, x + node * TLEN + tid * 4);
  for (int i = tid; i < HIDN * 32; i += 256) {
    int ch = i >> 5, k = i & 31;
    wk[i] = (k < 25) ? (_Float16)cw[ch * 25 + k] : (_Float16)0.0f;
  }
  if (tid < HIDN) {
    float s = bg[tid] * rsqrtf(brv[tid] + EPSV);
    sc[tid] = s;
    sh[tid] = (cb[tid] - brm[tid]) * s + bb[tid];
  }
  wait_async0();
  __syncthreads();
  for (int i = tid; i < 544; i += 256) {
    int t = i - 12;
    xpad[i] = (t >= 0 && t < TLEN) ? (_Float16)xraw[t] : (_Float16)0.0f;
  }
  __syncthreads();

  int wave = tid >> 5, lane = tid & 31;
  int hh = lane >> 4, ncol = lane & 15, kb = hh * 16;
  int ct = wave;  // channel tile 0..7 -> channels ct*16..ct*16+15
  v16h aF = frag_a_f16(wk + ct * 16 * 32, 32);
  float msc[8], msh[8], acc[8];
#pragma unroll
  for (int i = 0; i < 8; ++i) {
    int ch = ct * 16 + hh * 8 + i;
    msc[i] = sc[ch];
    msh[i] = sh[ch];
    acc[i] = 0.f;
  }
  for (int tt = 0; tt < 32; ++tt) {      // 32 time tiles cover t 0..511
    int n0 = tt * 16;
    v16h bF;                              // B[k,n] = xpad[(n0+n) + k]
#pragma unroll
    for (int e = 0; e < 16; ++e) bF[e] = xpad[n0 + ncol + kb + e];
    v8f c = {};
    c = wmma16x16x32(aF, bF, c);
    float keep = ((n0 + ncol) < 498) ? 1.f : 0.f;  // avgpool+mean == mean t<498
#pragma unroll
    for (int i = 0; i < 8; ++i) {
      float v = c[i] * msc[i] + msh[i];
      v = v > 0.f ? v : 0.f;
      acc[i] += keep * v;
    }
  }
#pragma unroll
  for (int i = 0; i < 8; ++i) {           // reduce over the 16 time columns
#pragma unroll
    for (int off = 1; off < 16; off <<= 1)
      acc[i] += __shfl_xor(acc[i], off, 16);
  }
  if (ncol == 0) {
#pragma unroll
    for (int i = 0; i < 8; ++i)
      feat[node * HIDN + ct * 16 + hh * 8 + i] = acc[i] * (1.0f / 498.0f);
  }
}

// ---------------- Setup kernels: degree, dinv, dense per-graph adjacency --
__global__ void k_init(float* __restrict__ Ahat, float* __restrict__ deg) {
  int i = blockIdx.x * blockDim.x + threadIdx.x;
  if (i < 64 * 64 * 64) Ahat[i] = 0.f;
  if (i < 4096) deg[i] = 1.0f;  // self loop weight 1
}
__global__ void k_deg(const int* __restrict__ ei, const float* __restrict__ ea,
                      float* __restrict__ deg, int E) {
  int e = blockIdx.x * blockDim.x + threadIdx.x;
  if (e < E) atomicAdd(&deg[ei[E + e]], ea[e]);
}
__global__ void k_dinv(float* __restrict__ deg) {
  int i = blockIdx.x * blockDim.x + threadIdx.x;
  if (i < 4096) deg[i] = rsqrtf(deg[i]);
}
__global__ void k_adj(const int* __restrict__ ei, const float* __restrict__ ea,
                      const int* __restrict__ batch,
                      const float* __restrict__ dinv, float* __restrict__ Ahat,
                      int E) {
  int e = blockIdx.x * blockDim.x + threadIdx.x;
  if (e >= E) return;
  int s = ei[e], d = ei[E + e];
  int g = batch[s];
  Ahat[g * 4096 + (d - g * 64) * 64 + (s - g * 64)] =
      dinv[s] * ea[e] * dinv[d];
}
__global__ void k_diag(const int* __restrict__ batch,
                       const float* __restrict__ dinv,
                       float* __restrict__ Ahat) {
  int n = blockIdx.x * blockDim.x + threadIdx.x;
  if (n >= 4096) return;
  int g = batch[n], l = n - g * 64;
  Ahat[g * 4096 + l * 64 + l] = dinv[n] * dinv[n];
}

// ---------------- Kernel 2: persistent per-graph GCN + pool + classifier --
// One block per graph. h kept in LDS (f32). Per layer:
//   hw = h @ W_l        (64x128 @ 128x128, WMMA, B streamed from global)
//   h  = relu(bn(Ahat_g @ hw + b))   (64x64 @ 64x128, WMMA)
// Then mean-pool + 2-layer MLP (scalar; negligible FLOPs).
__global__ __launch_bounds__(256) void k_gnn(
    const float* __restrict__ feat, const float* __restrict__ Ahat,
    const float* __restrict__ gw, const float* __restrict__ gb,
    const float* __restrict__ bng, const float* __restrict__ bnb,
    const float* __restrict__ bnrm, const float* __restrict__ bnrv,
    const float* __restrict__ w1, const float* __restrict__ b1,
    const float* __restrict__ w2, const float* __restrict__ b2,
    float* __restrict__ out) {
  __shared__ __align__(16) float hbuf[64 * 128];     // 32 KB f32 activations
  __shared__ __align__(16) _Float16 hwB[64 * 128];   // 16 KB; also Ahat stage
  __shared__ _Float16 Af[64 * 72];                   // 9 KB padded adjacency
  __shared__ float pooled[128];
  __shared__ float zc[64];
  int tid = threadIdx.x, g = blockIdx.x;
  // async DMA: feat tile (32 KB) -> hbuf ; Ahat tile (16 KB) -> hwB region
  float* stage = (float*)hwB;  // hwB is dead until GEMM1, reuse as f32 stage
  for (int c = tid; c < 2048; c += 256)
    async_copy16(hbuf + c * 4, feat + (size_t)g * 8192 + c * 4);
  for (int c = tid; c < 1024; c += 256)
    async_copy16(stage + c * 4, Ahat + (size_t)g * 4096 + c * 4);
  wait_async0();
  __syncthreads();
  for (int i = tid; i < 64 * 64; i += 256)
    Af[(i >> 6) * 72 + (i & 63)] = (_Float16)stage[i];

  int wave = tid >> 5, lane = tid & 31;
  int hh = lane >> 4, ncol = lane & 15, rb = hh * 8;
  int nt = wave;  // output column tile 0..7

  for (int l = 0; l < 3; ++l) {
    const float* W = gw + l * HIDN * HIDN;
    if (l + 1 < 3)  // warm L2 for next layer's weights (global_prefetch_b8)
      __builtin_prefetch(gw + (l + 1) * HIDN * HIDN + tid * 64, 0, 3);
    __syncthreads();  // Af-stage reads done; h ready -> GEMM1 may write hwB
    // GEMM1: hw = h @ W
    for (int mt = 0; mt < 4; ++mt) {
      v8f acc = {};
#pragma unroll
      for (int ks = 0; ks < 4; ++ks) {
        v16h aF = frag_a_f32(hbuf + (mt * 16) * 128 + ks * 32, 128);
        v16h bF = frag_b_f32(W + (ks * 32) * 128 + nt * 16, 128);
        acc = wmma16x16x32(aF, bF, acc);
      }
      int col = nt * 16 + ncol;
#pragma unroll
      for (int i = 0; i < 8; ++i)
        hwB[(mt * 16 + rb + i) * 128 + col] = (_Float16)acc[i];
    }
    __syncthreads();
    // GEMM2: h = relu(bn(Ahat @ hw + b)); channel is fixed per lane
    int ch = nt * 16 + ncol;
    float bi = gb[l * HIDN + ch];
    float s = bng[l * HIDN + ch] * rsqrtf(bnrv[l * HIDN + ch] + EPSV);
    float sb = bnb[l * HIDN + ch] - bnrm[l * HIDN + ch] * s;
    for (int mt = 0; mt < 4; ++mt) {
      v8f acc = {};
#pragma unroll
      for (int ks = 0; ks < 2; ++ks) {
        v16h aF = frag_a_f16(Af + (mt * 16) * 72 + ks * 32, 72);
        v16h bF = frag_b_f16(hwB + (ks * 32) * 128 + nt * 16, 128);
        acc = wmma16x16x32(aF, bF, acc);
      }
#pragma unroll
      for (int i = 0; i < 8; ++i) {
        float v = (acc[i] + bi) * s + sb;
        hbuf[(mt * 16 + rb + i) * 128 + ch] = v > 0.f ? v : 0.f;
      }
    }
  }
  __syncthreads();
  if (tid < 128) {  // global mean pool (each graph has exactly 64 nodes)
    float sum = 0.f;
    for (int r = 0; r < 64; ++r) sum += hbuf[r * 128 + tid];
    pooled[tid] = sum * (1.f / 64.f);
  }
  __syncthreads();
  if (tid < 64) {   // z = relu(pooled @ w1 + b1)
    float z = b1[tid];
    for (int c = 0; c < 128; ++c) z += pooled[c] * w1[c * 64 + tid];
    zc[tid] = z > 0.f ? z : 0.f;
  }
  __syncthreads();
  if (tid < 4) {    // out = z @ w2 + b2
    float o = b2[tid];
    for (int j = 0; j < 64; ++j) o += zc[j] * w2[j * 4 + tid];
    out[g * 4 + tid] = o;
  }
}

// ---------------------------------------------------------------------------
extern "C" void kernel_launch(void* const* d_in, const int* in_sizes, int n_in,
                              void* d_out, int out_size, void* d_ws,
                              size_t ws_size, hipStream_t stream) {
  const float* x    = (const float*)d_in[0];
  const int*   ei   = (const int*)d_in[1];
  const float* ea   = (const float*)d_in[2];
  const int*   bat  = (const int*)d_in[3];
  // d_in[4] = num_graphs (64, implied by layout)
  const float* cw   = (const float*)d_in[5];
  const float* cb   = (const float*)d_in[6];
  const float* bntg = (const float*)d_in[7];
  const float* bntb = (const float*)d_in[8];
  const float* bntrm= (const float*)d_in[9];
  const float* bntrv= (const float*)d_in[10];
  const float* gw   = (const float*)d_in[11];
  const float* gb   = (const float*)d_in[12];
  const float* bng  = (const float*)d_in[13];
  const float* bnb  = (const float*)d_in[14];
  const float* bnrm = (const float*)d_in[15];
  const float* bnrv = (const float*)d_in[16];
  const float* w1   = (const float*)d_in[17];
  const float* b1   = (const float*)d_in[18];
  const float* w2   = (const float*)d_in[19];
  const float* b2   = (const float*)d_in[20];
  int E = in_sizes[2];

  float* ws   = (float*)d_ws;
  float* feat = ws;                 // 4096*128 f32  (2 MB)
  float* deg  = ws + 4096 * 128;    // 4096 f32 (deg, then dinv in place)
  float* Ahat = deg + 4096;         // 64*64*64 f32  (1 MB)

  k_init<<<1024, 256, 0, stream>>>(Ahat, deg);
  k_conv<<<4096, 256, 0, stream>>>(x, cw, cb, bntg, bntb, bntrm, bntrv, feat);
  k_deg<<<(E + 255) / 256, 256, 0, stream>>>(ei, ea, deg, E);
  k_dinv<<<16, 256, 0, stream>>>(deg);
  k_adj<<<(E + 255) / 256, 256, 0, stream>>>(ei, ea, bat, deg, Ahat, E);
  k_diag<<<16, 256, 0, stream>>>(bat, deg, Ahat);
  k_gnn<<<64, 256, 0, stream>>>(feat, Ahat, gw, gb, bng, bnb, bnrm, bnrv,
                                w1, b1, w2, b2, (float*)d_out);
}
